// ImplicitDependencyRepresentations_75170517614653
// MI455X (gfx1250) — matrix operand order repaired and checked by reading, not compiled, not run on hardware
//
#include <hip/hip_runtime.h>
#include <hip/hip_bf16.h>
#include <math.h>

// Problem constants (from reference)
#define BB   32
#define LL   128
#define H2   800
#define ARC  500
#define ARCP 512          // ARC zero-padded to a multiple of 32
#define REL  100
#define NREL 45
#define MTOT (BB * LL)    // 4096 tokens
#define NCAT 1200         // [x_dep(600) | x_head(600)]

typedef __attribute__((ext_vector_type(2)))  float  v2f;
typedef __attribute__((ext_vector_type(8)))  float  v8f;
typedef __attribute__((ext_vector_type(8)))  __bf16 v8bf;
typedef __attribute__((ext_vector_type(16))) __bf16 v16bf;

union ABf { v16bf v; v8bf c[2]; };

// ---- WMMA wrappers --------------------------------------------------------
// fp32 path: D(16x16) = A(16x4) * B(4x16) + C
//  A/B: lane l, VGPR v -> (M|N) = l%16, K = v + (l>=16 ? 2 : 0)
//  C/D: lane l, VGPR r -> N = l%16, M = r + (l>=16 ? 8 : 0)
__device__ __forceinline__ v8f wmma_f32_4(v2f a, v2f b, v8f c) {
  return __builtin_amdgcn_wmma_f32_16x16x4_f32(false, a, false, b, (short)0, c,
                                               false, false);
}
// bf16 path: D(16x16,f32) = A(16x32,bf16) * B(32x16,bf16) + C
__device__ __forceinline__ v8f wmma_bf16(v16bf a, v16bf b, v8f c) {
  return __builtin_amdgcn_wmma_f32_16x16x32_bf16(false, a, false, b, (short)0, c,
                                                 false, false);
}

// A 16x32 bf16 fragment (ISA 7.12.2): half=0 lanes carry K 0..7 & 16..23,
// half=1 lanes carry K 8..15 & 24..31 -> two contiguous 16B chunks.
__device__ __forceinline__ v16bf load_afrag(const __bf16* rowk, int half) {
  ABf u;
  u.c[0] = *(const v8bf*)(rowk + half * 8);
  u.c[1] = *(const v8bf*)(rowk + 16 + half * 8);
  return u.v;
}
// B 32x16 bf16 fragment from K-major (transposed) storage: half=0 -> K 0..7 & 16..23,
// half=1 -> K 8..15 & 24..31 -> two contiguous 16B chunks (matches A).
__device__ __forceinline__ v16bf load_bfrag(const __bf16* colk, int half) {
  ABf u;
  u.c[0] = *(const v8bf*)(colk + half * 8);
  u.c[1] = *(const v8bf*)(colk + 16 + half * 8);
  return u.v;
}

// fp32 -> bf16 hi/lo split (hi = exact top 16 bits, lo = RTNE residual)
__device__ __forceinline__ void bfsplit(float v, __bf16* h, __bf16* l) {
  const float fh = __uint_as_float(__float_as_uint(v) & 0xffff0000u);
  *h = (__bf16)fh;
  *l = (__bf16)(v - fh);
}

// ---------------------------------------------------------------------------
// Conversion kernels (operand packing for the bf16x3 GEMMs)
__global__ void k_cvt_x(const float* __restrict__ X, __bf16* __restrict__ Xhi,
                        __bf16* __restrict__ Xlo) {
  const int i = blockIdx.x * 256 + threadIdx.x;      // MTOT*H2 exact
  bfsplit(X[i], Xhi + i, Xlo + i);
}

// W concat + transpose: Wt[n, k] = (n<600 ? W_dep : W_head)[k, n%600]
__global__ void k_cvt_w(const float* __restrict__ Wd, const float* __restrict__ Wh,
                        __bf16* __restrict__ Wthi, __bf16* __restrict__ Wtlo) {
  const int i = blockIdx.x * 256 + threadIdx.x;      // NCAT*H2 exact
  const int n = i / H2, k = i % H2;
  const float v = (n < 600) ? Wd[(size_t)k * 600 + n]
                            : Wh[(size_t)k * 600 + (n - 600)];
  bfsplit(v, Wthi + i, Wtlo + i);
}

// arc_W transpose + zero-pad to 512x512: AWt[n, k] = arc_W[k, n]
__global__ void k_cvt_arcw(const float* __restrict__ arcW,
                           __bf16* __restrict__ AWhi, __bf16* __restrict__ AWlo) {
  const int i = blockIdx.x * 256 + threadIdx.x;      // ARCP*ARCP exact
  const int n = i >> 9, k = i & (ARCP - 1);
  const float v = (n < ARC && k < ARC) ? arcW[(size_t)k * ARC + n] : 0.f;
  bfsplit(v, AWhi + i, AWlo + i);
}

// x_arc_dep rows, zero-padded to K=512
__global__ void k_cvt_arcdep(const float* __restrict__ xfeat,
                             __bf16* __restrict__ Adhi, __bf16* __restrict__ Adlo) {
  const int i = blockIdx.x * 256 + threadIdx.x;      // MTOT*ARCP exact
  const int m = i >> 9, k = i & (ARCP - 1);
  const float v = (k < ARC) ? xfeat[(size_t)m * NCAT + k] : 0.f;
  bfsplit(v, Adhi + i, Adlo + i);
}

// ---------------------------------------------------------------------------
// K1 (bf16x3): xfeat = leaky([lstm@W_dep+b_dep | lstm@W_head+b_head])
// M=4096, N=1200, K=800. Wave tile = 32x48 (2x3 WMMA subtiles, 6 f32 accs).
__global__ void k_proj_bf16(const __bf16* __restrict__ Xhi, const __bf16* __restrict__ Xlo,
                            const __bf16* __restrict__ Wthi, const __bf16* __restrict__ Wtlo,
                            const float* __restrict__ bd, const float* __restrict__ bh,
                            float* __restrict__ xfeat) {
  const int lane = threadIdx.x & 31;
  const int wid  = blockIdx.x * 8 + (threadIdx.x >> 5);  // 0..3199
  const int mb = wid / 25, nb = wid % 25;
  const int lm = lane & 15, half = lane >> 4;
  const int m0 = mb * 32, n0 = nb * 48;

  const __bf16 *ah[2], *al[2], *bh_[3], *bl[3];
#pragma unroll
  for (int ms = 0; ms < 2; ++ms) {
    ah[ms] = Xhi + (size_t)(m0 + ms * 16 + lm) * H2;
    al[ms] = Xlo + (size_t)(m0 + ms * 16 + lm) * H2;
  }
#pragma unroll
  for (int ns = 0; ns < 3; ++ns) {
    const int n = n0 + ns * 16 + lm;
    bh_[ns] = Wthi + (size_t)n * H2;
    bl[ns]  = Wtlo + (size_t)n * H2;
  }

  v8f acc[2][3] = {};
  for (int k = 0; k < H2; k += 32) {
    __builtin_prefetch(bh_[0] + k + 128, 0, 1);   // global_prefetch_b8
    v16bf A[2], Al[2], Bm[3], Bl[3];
#pragma unroll
    for (int ms = 0; ms < 2; ++ms) {
      A[ms]  = load_afrag(ah[ms] + k, half);
      Al[ms] = load_afrag(al[ms] + k, half);
    }
#pragma unroll
    for (int ns = 0; ns < 3; ++ns) {
      Bm[ns] = load_bfrag(bh_[ns] + k, half);
      Bl[ns] = load_bfrag(bl[ns] + k, half);
    }
#pragma unroll
    for (int ms = 0; ms < 2; ++ms)
#pragma unroll
      for (int ns = 0; ns < 3; ++ns) {
        acc[ms][ns] = wmma_bf16(A[ms],  Bm[ns], acc[ms][ns]);
        acc[ms][ns] = wmma_bf16(A[ms],  Bl[ns], acc[ms][ns]);
        acc[ms][ns] = wmma_bf16(Al[ms], Bm[ns], acc[ms][ns]);
      }
  }

#pragma unroll
  for (int ms = 0; ms < 2; ++ms)
#pragma unroll
    for (int ns = 0; ns < 3; ++ns) {
      const int n = n0 + ns * 16 + lm;
      const float bias = (n < 600) ? bd[n] : bh[n - 600];
      float* orow = xfeat + (size_t)(m0 + ms * 16 + half * 8) * NCAT + n;
#pragma unroll
      for (int r = 0; r < 8; ++r) {
        float v = acc[ms][ns][r] + bias;
        orow[(size_t)r * NCAT] = (v > 0.f) ? v : 0.1f * v;
      }
    }
}

// ---------------------------------------------------------------------------
// K2 (bf16x3): tmp[m,j] = x_arc_dep[m,:]@arc_W[:500,j] + arc_W[500,j]
// M=4096, N=512(padded), K=512(padded). Wave tile = 32x32 (2x2 subtiles).
__global__ void k_arc_tmp_bf16(const __bf16* __restrict__ Adhi, const __bf16* __restrict__ Adlo,
                               const __bf16* __restrict__ AWhi, const __bf16* __restrict__ AWlo,
                               const float* __restrict__ arcW,
                               float* __restrict__ tmp) {
  const int lane = threadIdx.x & 31;
  const int wid  = blockIdx.x * 8 + (threadIdx.x >> 5);  // 0..2047
  const int mb = wid >> 4, nb = wid & 15;
  const int lm = lane & 15, half = lane >> 4;
  const int m0 = mb * 32, n0 = nb * 32;

  const __bf16 *ah[2], *al[2], *bh_[2], *bl[2];
#pragma unroll
  for (int ms = 0; ms < 2; ++ms) {
    ah[ms] = Adhi + (size_t)(m0 + ms * 16 + lm) * ARCP;
    al[ms] = Adlo + (size_t)(m0 + ms * 16 + lm) * ARCP;
  }
#pragma unroll
  for (int ns = 0; ns < 2; ++ns) {
    const int n = n0 + ns * 16 + lm;
    bh_[ns] = AWhi + (size_t)n * ARCP;
    bl[ns]  = AWlo + (size_t)n * ARCP;
  }

  v8f acc[2][2] = {};
  for (int k = 0; k < ARCP; k += 32) {
    v16bf A[2], Al[2], Bm[2], Bl[2];
#pragma unroll
    for (int ms = 0; ms < 2; ++ms) {
      A[ms]  = load_afrag(ah[ms] + k, half);
      Al[ms] = load_afrag(al[ms] + k, half);
    }
#pragma unroll
    for (int ns = 0; ns < 2; ++ns) {
      Bm[ns] = load_bfrag(bh_[ns] + k, half);
      Bl[ns] = load_bfrag(bl[ns] + k, half);
    }
#pragma unroll
    for (int ms = 0; ms < 2; ++ms)
#pragma unroll
      for (int ns = 0; ns < 2; ++ns) {
        acc[ms][ns] = wmma_bf16(A[ms],  Bm[ns], acc[ms][ns]);
        acc[ms][ns] = wmma_bf16(A[ms],  Bl[ns], acc[ms][ns]);
        acc[ms][ns] = wmma_bf16(Al[ms], Bm[ns], acc[ms][ns]);
      }
  }

#pragma unroll
  for (int ms = 0; ms < 2; ++ms)
#pragma unroll
    for (int ns = 0; ns < 2; ++ns) {
      const int n = n0 + ns * 16 + lm;
      if (n < ARC) {
        const float bias = arcW[(size_t)ARC * ARC + n];   // ones-column row
        float* orow = tmp + (size_t)(m0 + ms * 16 + half * 8) * ARC + n;
#pragma unroll
        for (int r = 0; r < 8; ++r) orow[(size_t)r * ARC] = acc[ms][ns][r] + bias;
      }
    }
}

// ---------------------------------------------------------------------------
// K3 (fp32 WMMA): arc_logits[b,x,y] = tmp[b*128+x,:] . x_arc_head[b*128+y,:]
__global__ void k_arc_logits(const float* __restrict__ tmp,
                             const float* __restrict__ xfeat,
                             float* __restrict__ arc) {
  const int lane = threadIdx.x & 31;
  const int yt   = threadIdx.x >> 5;
  const int xt   = blockIdx.x;
  const int b    = blockIdx.y;
  const int lm   = lane & 15;
  const int koff = (lane >> 4) * 2;

  const float* arow = tmp + (size_t)(b * LL + xt * 16 + lm) * ARC;
  const float* brow = xfeat + (size_t)(b * LL + yt * 16 + lm) * NCAT + 600;

  v8f acc = {};
#pragma unroll 4
  for (int k = 0; k < ARC; k += 4) {
    v2f a, bb;
    a.x  = arow[k + koff];
    a.y  = arow[k + koff + 1];
    bb.x = brow[k + koff];
    bb.y = brow[k + koff + 1];
    acc = wmma_f32_4(a, bb, acc);
  }
  float* orow = arc + (size_t)(b * LL + xt * 16 + (lane >> 4) * 8) * LL + yt * 16 + lm;
#pragma unroll
  for (int r = 0; r < 8; ++r) orow[(size_t)r * LL] = acc[r];
}

// ---------------------------------------------------------------------------
// K4: rel loss (per-token bilinear + CE, gather on heads)
__global__ void k_rel_loss(const float* __restrict__ xfeat,
                           const int* __restrict__ heads,
                           const int* __restrict__ rels,
                           const float* __restrict__ relW,
                           float* __restrict__ loss_acc) {
  const int t = blockIdx.x;
  const int lbl = rels[t];
  if (lbl < 0) return;                    // block-uniform, before any barrier
  const int b = t >> 7;
  int head = heads[t]; if (head < 0) head = 0;
  const int tid = threadIdx.x;

  __shared__ float p[104], q[104];
  __shared__ float h[NREL * 101];
  __shared__ float logits[48];

  for (int i = tid; i < 101; i += 128)
    p[i] = (i < REL) ? xfeat[(size_t)t * NCAT + 500 + i] : 1.0f;
  const int qrow = b * LL + head;
  for (int j = tid; j < 101; j += 128)
    q[j] = (j < REL) ? xfeat[(size_t)qrow * NCAT + 1100 + j] : 1.0f;
  __syncthreads();

  for (int pair = tid; pair < NREL * 101; pair += 128) {  // pair = o*101 + i
    const float* w = relW + (size_t)pair * 101;
    float s = 0.f;
    for (int j = 0; j < 101; ++j) s += w[j] * q[j];
    h[pair] = s;
  }
  __syncthreads();

  for (int o = tid; o < NREL; o += 128) {
    const float* hr = h + o * 101;
    float s = 0.f;
    for (int i = 0; i < 101; ++i) s += p[i] * hr[i];
    logits[o] = s;
  }
  __syncthreads();

  if (tid == 0) {
    float m = -1e30f;
    for (int o = 0; o < NREL; ++o) m = fmaxf(m, logits[o]);
    float s = 0.f;
    for (int o = 0; o < NREL; ++o) s += __expf(logits[o] - m);
    atomicAdd(loss_acc, -(logits[lbl] - m - __logf(s)));
  }
}

// ---------------------------------------------------------------------------
// K5: arc loss. One wave per token: CE over 128 masked logits vs heads[t].
__global__ void k_arc_loss(const float* __restrict__ arc,
                           const int* __restrict__ heads,
                           const int* __restrict__ slen,
                           float* __restrict__ loss_acc) {
  const int lane = threadIdx.x & 31;
  const int wave = threadIdx.x >> 5;
  const int t = blockIdx.x * 8 + wave;
  const int lbl = heads[t];
  if (lbl < 0) return;                    // wave-uniform
  const int len = slen[t >> 7];
  const float* row = arc + (size_t)t * LL;

  float v[4], m = -1e30f;
#pragma unroll
  for (int i = 0; i < 4; ++i) {
    const int y = lane + 32 * i;
    v[i] = row[y] + ((y < len) ? 0.f : -1000.f);
    m = fmaxf(m, v[i]);
  }
  for (int off = 16; off; off >>= 1) m = fmaxf(m, __shfl_xor(m, off, 32));
  float s = 0.f;
#pragma unroll
  for (int i = 0; i < 4; ++i) s += __expf(v[i] - m);
  for (int off = 16; off; off >>= 1) s += __shfl_xor(s, off, 32);
  if (lane == 0) {
    const float lg = row[lbl] + ((lbl < len) ? 0.f : -1000.f);
    atomicAdd(loss_acc, -(lg - m - __logf(s)));
  }
}

__global__ void k_zero(float* p) { if (threadIdx.x == 0) p[0] = 0.f; }
__global__ void k_final(const float* loss, float* out) {
  if (threadIdx.x == 0) out[0] = loss[0];
}

// ---------------------------------------------------------------------------
extern "C" void kernel_launch(void* const* d_in, const int* in_sizes, int n_in,
                              void* d_out, int out_size, void* d_ws, size_t ws_size,
                              hipStream_t stream) {
  const float* lstm  = (const float*)d_in[0];
  const int*   slen  = (const int*)d_in[1];
  const int*   heads = (const int*)d_in[2];
  const int*   rels  = (const int*)d_in[3];
  const float* Wd    = (const float*)d_in[4];
  const float* bd    = (const float*)d_in[5];
  const float* Wh    = (const float*)d_in[6];
  const float* bh    = (const float*)d_in[7];
  const float* arcW  = (const float*)d_in[8];
  const float* relW  = (const float*)d_in[9];

  float* out = (float*)d_out;
  float* arc = out + 1;                                  // (B,L,L) logits

  // Workspace carve-up (all 256B aligned)
  char* ws = (char*)d_ws;
  size_t off = 0;
  float* loss = (float*)(ws + off);  off += 256;
  float* xfeat = (float*)(ws + off); off += (size_t)MTOT * NCAT * 4;   // 19.7 MB
  float* tmp   = (float*)(ws + off); off += (size_t)MTOT * ARC * 4;    //  8.2 MB
  __bf16* Xhi  = (__bf16*)(ws + off); off += (size_t)MTOT * H2 * 2;    //  6.6 MB
  __bf16* Xlo  = (__bf16*)(ws + off); off += (size_t)MTOT * H2 * 2;
  __bf16* Wthi = (__bf16*)(ws + off); off += (size_t)NCAT * H2 * 2;    //  1.9 MB
  __bf16* Wtlo = (__bf16*)(ws + off); off += (size_t)NCAT * H2 * 2;
  __bf16* Adhi = (__bf16*)(ws + off); off += (size_t)MTOT * ARCP * 2;  //  4.2 MB
  __bf16* Adlo = (__bf16*)(ws + off); off += (size_t)MTOT * ARCP * 2;
  __bf16* AWhi = (__bf16*)(ws + off); off += (size_t)ARCP * ARCP * 2;  //  0.5 MB
  __bf16* AWlo = (__bf16*)(ws + off); off += (size_t)ARCP * ARCP * 2;

  k_zero<<<1, 64, 0, stream>>>(loss);

  // Operand packing (hi/lo bf16 splits, transposes, padding)
  k_cvt_x   <<<(MTOT * H2) / 256, 256, 0, stream>>>(lstm, Xhi, Xlo);
  k_cvt_w   <<<(NCAT * H2) / 256, 256, 0, stream>>>(Wd, Wh, Wthi, Wtlo);
  k_cvt_arcw<<<(ARCP * ARCP) / 256, 256, 0, stream>>>(arcW, AWhi, AWlo);

  // K1: 128 m-blocks x 25 n-blocks, 8 waves/block
  k_proj_bf16<<<400, 256, 0, stream>>>(Xhi, Xlo, Wthi, Wtlo, bd, bh, xfeat);

  // Pack x_arc_dep (padded K=512) for K2
  k_cvt_arcdep<<<(MTOT * ARCP) / 256, 256, 0, stream>>>(xfeat, Adhi, Adlo);

  // K2: 128 m-blocks x 16 n-blocks, 8 waves/block
  k_arc_tmp_bf16<<<256, 256, 0, stream>>>(Adhi, Adlo, AWhi, AWlo, arcW, tmp);

  // K3: 8 x-tiles, 32 batches; 8 waves/block cover the 8 y-tiles (fp32 WMMA)
  k_arc_logits<<<dim3(8, 32), 256, 0, stream>>>(tmp, xfeat, arc);

  // K5: arc CE (reads arc logits from d_out)
  k_arc_loss<<<512, 256, 0, stream>>>(arc, heads, slen, loss);

  // K4: rel CE, one block per token
  k_rel_loss<<<MTOT, 128, 0, stream>>>(xfeat, heads, rels, relW, loss);

  k_final<<<1, 64, 0, stream>>>(loss, out);
}
